// MultiLayerAttention_40166534152720
// MI455X (gfx1250) — matrix operand order repaired and checked by reading, not compiled, not run on hardware
//
#include <hip/hip_runtime.h>

// ---------------------------------------------------------------------------
// MultiLayerAttention for MI455X (gfx1250): bf16 WMMA attention + FC + LN.
// N=8, S=1024, E=256, H=8, D=32, L=4.  d_out = [out (8,1024,256) | aw (8,8,1024,1024)]
// ---------------------------------------------------------------------------

typedef __attribute__((ext_vector_type(16))) __bf16 v16bf;
typedef __attribute__((ext_vector_type(8)))  float  v8f;
typedef __attribute__((ext_vector_type(4)))  unsigned int u32x4;
typedef int b128_t __attribute__((vector_size(16)));   // async-copy payload type

#define NB   8
#define SS   1024
#define EE   256
#define HH   8
#define DD   32
#define LL   4
#define SCALE 0.0625f   // 1/sqrt(256)

__device__ __forceinline__ unsigned pack_bf16(float x, float y) {
  __bf16 a = (__bf16)x, b = (__bf16)y;
  unsigned short ua = __builtin_bit_cast(unsigned short, a);
  unsigned short ub = __builtin_bit_cast(unsigned short, b);
  return (unsigned)ua | ((unsigned)ub << 16);
}

// ---- gfx1250 async global->LDS copy (ASYNCcnt-tracked), with safe fallback -
__device__ __forceinline__ void async_copy_b128(const __bf16* g, __bf16* l) {
#if __has_builtin(__builtin_amdgcn_global_load_async_to_lds_b128)
  __builtin_amdgcn_global_load_async_to_lds_b128(
      (b128_t*)g, (b128_t*)l, 0, 0);
#else
  *(u32x4*)l = *(const u32x4*)g;
#endif
}

__device__ __forceinline__ void wait_async() {
#if __has_builtin(__builtin_amdgcn_global_load_async_to_lds_b128)
#if __has_builtin(__builtin_amdgcn_s_wait_asynccnt)
  __builtin_amdgcn_s_wait_asynccnt(0);
#else
  asm volatile("s_wait_asynccnt 0x0" ::: "memory");
#endif
#endif
}

// ---------------------------------------------------------------------------
// Kernel 0: convert K -> bf16 [N,H,S,D], V -> bf16 transposed [N,H,D,S]
// ---------------------------------------------------------------------------
__global__ void convert_kv(const float* __restrict__ values,
                           const float* __restrict__ keys,
                           __bf16* __restrict__ Kbf,
                           __bf16* __restrict__ Vt) {
  size_t idx = (size_t)blockIdx.x * 256 + threadIdx.x;
  if (idx >= (size_t)NB * SS * EE) return;
  int e = (int)(idx & (EE - 1));
  size_t t = idx >> 8;            // token index
  int s = (int)(t & (SS - 1));
  int n = (int)(t >> 10);
  int h = e >> 5, d = e & 31;
  int nh = n * HH + h;
  Kbf[((size_t)nh * SS + s) * DD + d] = (__bf16)keys[idx];
  Vt [((size_t)nh * DD + d) * SS + s] = (__bf16)values[idx];
}

// Kernel 0b: convert all layer FC weights to bf16 (same [L,E,E] layout)
__global__ void convert_w(const float* __restrict__ W, __bf16* __restrict__ Wbf) {
  size_t idx = (size_t)blockIdx.x * 256 + threadIdx.x;
  if (idx < (size_t)LL * EE * EE) Wbf[idx] = (__bf16)W[idx];
}

// ---------------------------------------------------------------------------
// Kernel A: attention for one (n,h) and 32 query rows per block.
//   64 threads = 2 waves; wave w owns query rows [qbase+16w, qbase+16w+16).
//   K is streamed through LDS in 64-key double-buffered async chunks.
// ---------------------------------------------------------------------------
__global__ void __launch_bounds__(64)
attn_kernel(const float* __restrict__ cur,      // [N,S,E] current layer input
            const __bf16* __restrict__ Kbf,     // [N,H,S,D]
            const __bf16* __restrict__ Vt,      // [N,H,D,S]
            float* __restrict__ oBuf,           // [N,S,E] attention output (pre-FC)
            float* __restrict__ awOut,          // [N,H,S,S] accumulated mean
            int firstLayer) {
  __shared__ float  en[32 * SS];        // 128 KB: energies, then normalized probs
  __shared__ __bf16 qs[32 * DD];        // 2 KB q tile (bf16)
  __shared__ __bf16 ks[2][64 * DD];     // 8 KB double-buffered K chunks
  __shared__ float  red[64];
  __shared__ float  rowmax[32];
  __shared__ float  rowrinv[32];

  const int tid  = threadIdx.x;
  const int nh   = blockIdx.x;          // 0..63
  const int n    = nh >> 3;
  const int h    = nh & 7;
  const int qbase = blockIdx.y * 32;

  // ---- stage q tile (32 rows x 32 cols), fp32 -> bf16 --------------------
  for (int i = tid; i < 32 * DD; i += 64) {
    int rr = i >> 5, cc = i & 31;
    qs[i] = (__bf16)cur[((size_t)n * SS + qbase + rr) * EE + h * DD + cc];
  }

  const __bf16* kbase = Kbf + (size_t)nh * SS * DD;

  // prime chunk 0 (each thread copies 64B = 4 x b128)
  {
    const __bf16* g = kbase + tid * 32;
    __bf16* l = &ks[0][tid * 32];
#pragma unroll
    for (int j = 0; j < 4; ++j) async_copy_b128(g + j * 8, l + j * 8);
  }
  __syncthreads();

  const int wave = tid >> 5, lane = tid & 31;
  const int half = lane >> 4, ln = lane & 15;
  const int m = ln + 16 * wave;         // this lane's A-matrix row (0..31)

  // ---- A fragment for Q (16x32 bf16, contiguous k = 16*half + s) -------------------------
  v16bf aq;
  {
    unsigned* ap = (unsigned*)&aq;
#pragma unroll
    for (int v = 0; v < 8; ++v) {
      int k0 = 2 * v + 16 * half;
      ap[v] = *(const unsigned*)&qs[m * DD + k0];
    }
  }

  // ---- energy = Q * K^T, K double-buffered through LDS -------------------
  for (int c = 0; c < SS / 64; ++c) {
    const int buf = c & 1;
    wait_async();          // this wave's chunk-c async writes complete
    __syncthreads();       // all waves' chunk-c data visible; prev buffer free
    if (c + 1 < SS / 64) {
      const __bf16* g = kbase + (size_t)(c + 1) * 64 * DD + tid * 32;
      __bf16* l = &ks[buf ^ 1][tid * 32];
#pragma unroll
      for (int j = 0; j < 4; ++j) async_copy_b128(g + j * 8, l + j * 8);
    }
#pragma unroll
    for (int t = 0; t < 4; ++t) {
      const int kl = t * 16 + ln;       // key index within chunk (B column)
      v16bf b;
      const u32x4* bp = (const u32x4*)&ks[buf][kl * DD + 16 * half];
      ((u32x4*)&b)[0] = bp[0];
      ((u32x4*)&b)[1] = bp[1];
      v8f acc = {};
      acc = __builtin_amdgcn_wmma_f32_16x16x32_bf16(false, aq, false, b,
                                                    (short)0, acc, false, false);
      const int kcol = c * 64 + kl;
#pragma unroll
      for (int r = 0; r < 8; ++r)
        en[(16 * wave + r + 8 * half) * SS + kcol] = acc[r];
    }
  }
  __syncthreads();

  // ---- softmax over each of 32 rows (length 1024) ------------------------
  {
    const int r = tid >> 1, c0 = (tid & 1) * 512;
    float mx = -3.4e38f;
    for (int c = 0; c < 512; ++c) mx = fmaxf(mx, en[r * SS + c0 + c]);
    red[tid] = mx;
    __syncthreads();
    if (!(tid & 1)) rowmax[r] = fmaxf(red[tid], red[tid + 1]);
    __syncthreads();
    const float rm = rowmax[r];
    float s = 0.f;
    for (int c = 0; c < 512; ++c) {
      float ev = __expf((en[r * SS + c0 + c] - rm) * SCALE);
      en[r * SS + c0 + c] = ev;
      s += ev;
    }
    red[tid] = s;
    __syncthreads();
    if (!(tid & 1)) rowrinv[r] = 1.0f / (red[tid] + red[tid + 1]);
    __syncthreads();
  }
  // normalize + accumulate aw/4 into d_out (coalesced)
  {
    float* awBase = awOut + ((size_t)nh * SS + qbase) * SS;
    for (int i = tid; i < 32 * SS; i += 64) {
      int rr = i >> 10;
      float p = en[i] * rowrinv[rr];
      en[i] = p;
      if (firstLayer) awBase[i] = 0.25f * p;
      else            awBase[i] += 0.25f * p;
    }
  }
  __syncthreads();

  // ---- o = aw * V  (Vt is [D,S] so B fragments are contiguous) -----------
  const __bf16* vtbase = Vt + (size_t)nh * DD * SS;
#pragma unroll
  for (int dt = 0; dt < 2; ++dt) {
    v8f acc = {};
    const int d = dt * 16 + ln;         // output feature column (B's N dim)
    const __bf16* vrow = vtbase + (size_t)d * SS;
    for (int kc = 0; kc < SS; kc += 32) {
      if (kc + 256 < SS)
        __builtin_prefetch(vrow + kc + 256, 0, 3);
      v16bf a;
      unsigned* ap = (unsigned*)&a;
#pragma unroll
      for (int v = 0; v < 8; ++v) {
        int k0 = 2 * v + 16 * half;
        float2 f = *(const float2*)&en[m * SS + kc + k0];
        ap[v] = pack_bf16(f.x, f.y);
      }
      v16bf b;
      const u32x4* bp = (const u32x4*)(vrow + kc + 16 * half);
      ((u32x4*)&b)[0] = bp[0];
      ((u32x4*)&b)[1] = bp[1];
      acc = __builtin_amdgcn_wmma_f32_16x16x32_bf16(false, a, false, b,
                                                    (short)0, acc, false, false);
    }
#pragma unroll
    for (int r = 0; r < 8; ++r) {
      int row = qbase + 16 * wave + r + 8 * half;
      oBuf[((size_t)n * SS + row) * EE + h * DD + dt * 16 + ln] = acc[r];
    }
  }
}

// ---------------------------------------------------------------------------
// Kernel B: out = LayerNorm(oBuf @ W^T + bias + cur); 32 tokens per block.
// ---------------------------------------------------------------------------
__global__ void __launch_bounds__(64)
fc_ln_kernel(const float* __restrict__ oBuf,
             const float* __restrict__ cur,
             const __bf16* __restrict__ Wbf,  // [E,E] bf16, row-major (out_e, in_k)
             const float* __restrict__ bias,
             const float* __restrict__ gam,
             const float* __restrict__ bet,
             float* __restrict__ dst) {
  __shared__ __bf16 obf[32 * EE];   // 16 KB bf16 A-tile staging
  __shared__ float  x[32 * EE];     // 32 KB FC result
  __shared__ float  red[64];
  __shared__ float  mu[32];
  __shared__ float  rvar[32];

  const int tid = threadIdx.x;
  const int rowbase = blockIdx.x * 32;

  for (int i = tid; i < 32 * EE; i += 64) {
    int rr = i >> 8, cc = i & 255;
    obf[i] = (__bf16)oBuf[(size_t)(rowbase + rr) * EE + cc];
  }
  __syncthreads();

  const int wave = tid >> 5, lane = tid & 31;
  const int half = lane >> 4, ln = lane & 15;
  const int m = ln + 16 * wave;

  for (int et = 0; et < 16; ++et) {
    v8f acc = {};
    const int e = et * 16 + ln;                 // B column = output feature
    const __bf16* wrow = Wbf + (size_t)e * EE;
    for (int kc = 0; kc < EE; kc += 32) {
      v16bf a;
      unsigned* ap = (unsigned*)&a;
#pragma unroll
      for (int v = 0; v < 8; ++v) {
        int k0 = 2 * v + 16 * half;
        ap[v] = *(const unsigned*)&obf[m * EE + kc + k0];
      }
      v16bf b;
      const u32x4* bp = (const u32x4*)(wrow + kc + 16 * half);
      ((u32x4*)&b)[0] = bp[0];
      ((u32x4*)&b)[1] = bp[1];
      acc = __builtin_amdgcn_wmma_f32_16x16x32_bf16(false, a, false, b,
                                                    (short)0, acc, false, false);
    }
#pragma unroll
    for (int r = 0; r < 8; ++r)
      x[(16 * wave + r + 8 * half) * EE + et * 16 + ln] = acc[r];
  }
  __syncthreads();

  // bias + residual
  for (int i = tid; i < 32 * EE; i += 64) {
    int rr = i >> 8, cc = i & 255;
    x[i] += bias[cc] + cur[(size_t)(rowbase + rr) * EE + cc];
  }
  __syncthreads();

  // LayerNorm per row
  const int r = tid >> 1, c0 = (tid & 1) * 128;
  float s = 0.f;
  for (int c = 0; c < 128; ++c) s += x[r * EE + c0 + c];
  red[tid] = s;
  __syncthreads();
  if (!(tid & 1)) mu[r] = (red[tid] + red[tid + 1]) * (1.0f / 256.0f);
  __syncthreads();
  const float mm = mu[r];
  float vv = 0.f;
  for (int c = 0; c < 128; ++c) {
    float dx = x[r * EE + c0 + c] - mm;
    vv += dx * dx;
  }
  red[tid] = vv;
  __syncthreads();
  if (!(tid & 1))
    rvar[r] = rsqrtf((red[tid] + red[tid + 1]) * (1.0f / 256.0f) + 1e-5f);
  __syncthreads();

  for (int i = tid; i < 32 * EE; i += 64) {
    int rr = i >> 8, cc = i & 255;
    dst[(size_t)(rowbase + rr) * EE + cc] =
        gam[cc] * (x[i] - mu[rr]) * rvar[rr] + bet[cc];
  }
}

// ---------------------------------------------------------------------------
extern "C" void kernel_launch(void* const* d_in, const int* in_sizes, int n_in,
                              void* d_out, int out_size, void* d_ws, size_t ws_size,
                              hipStream_t stream) {
  (void)in_sizes; (void)n_in; (void)out_size; (void)ws_size;
  const float* values = (const float*)d_in[0];
  const float* keys   = (const float*)d_in[1];
  const float* query  = (const float*)d_in[2];
  const float* fc_w   = (const float*)d_in[3];
  const float* fc_b   = (const float*)d_in[4];
  const float* ln_g   = (const float*)d_in[5];
  const float* ln_b   = (const float*)d_in[6];

  float* out = (float*)d_out;                       // (8,1024,256)
  float* aw  = out + (size_t)NB * SS * EE;          // (8,8,1024,1024)

  char* ws = (char*)d_ws;
  __bf16* Kbf = (__bf16*)ws;                        // 4 MB
  __bf16* Vt  = (__bf16*)(ws + ((size_t)4 << 20));  // 4 MB
  float* oBuf = (float*)(ws + ((size_t)8 << 20));   // 8 MB
  float* bufA = (float*)(ws + ((size_t)16 << 20));  // 8 MB
  float* bufB = (float*)(ws + ((size_t)24 << 20));  // 8 MB
  __bf16* Wbf = (__bf16*)(ws + ((size_t)32 << 20)); // 512 KB

  {
    size_t total = (size_t)NB * SS * EE;
    convert_kv<<<(unsigned)((total + 255) / 256), 256, 0, stream>>>(values, keys, Kbf, Vt);
    size_t wtotal = (size_t)LL * EE * EE;
    convert_w<<<(unsigned)((wtotal + 255) / 256), 256, 0, stream>>>(fc_w, Wbf);
  }

  const float* curp = query;
  for (int l = 0; l < LL; ++l) {
    dim3 grid(NB * HH, SS / 32);
    attn_kernel<<<grid, 64, 0, stream>>>(curp, Kbf, Vt, oBuf, aw, l == 0 ? 1 : 0);
    float* dst = (l == LL - 1) ? out : ((l & 1) ? bufB : bufA);
    fc_ln_kernel<<<NB * SS / 32, 64, 0, stream>>>(
        oBuf, curp, Wbf + (size_t)l * EE * EE, fc_b + (size_t)l * EE,
        ln_g + (size_t)l * EE, ln_b + (size_t)l * EE, dst);
    curp = dst;
  }
}
